// MultiHeadAttention_8735963480358
// MI455X (gfx1250) — matrix operand order repaired and checked by reading, not compiled, not run on hardware
//
#include <hip/hip_runtime.h>
#include <stdint.h>

// ---------- types for CDNA5 WMMA ----------
typedef __attribute__((ext_vector_type(16))) __bf16 v16bf;
typedef __attribute__((ext_vector_type(8)))  float  v8f;
typedef __attribute__((ext_vector_type(4)))  float  v4f;
typedef __attribute__((ext_vector_type(8)))  unsigned short v8us;
typedef __attribute__((ext_vector_type(4)))  int    v4i;

union BFrag {
    v16bf v;
    v8us  h[2];
    unsigned short s[16];
};

// ---------- gfx1250 async global->LDS path (builtin confirmed present) ----------
#if __has_builtin(__builtin_amdgcn_global_load_async_to_lds_b128) && \
    __has_builtin(__builtin_amdgcn_s_wait_asynccnt)
#define HAVE_ASYNC_LDS 1
typedef __attribute__((address_space(1))) v4i* gas_v4i;
typedef __attribute__((address_space(3))) v4i* las_v4i;
#else
#define HAVE_ASYNC_LDS 0
#endif

__device__ __forceinline__ void stage16(const unsigned short* g, unsigned short* l) {
#if HAVE_ASYNC_LDS
    __builtin_amdgcn_global_load_async_to_lds_b128(
        (gas_v4i)(v4i*)g, (las_v4i)(v4i*)l, 0, 0);
#else
    *(v8us*)l = *(const v8us*)g;
#endif
}

__device__ __forceinline__ void wait_async0() {
#if HAVE_ASYNC_LDS
    __builtin_amdgcn_s_wait_asynccnt(0);
#endif
}

__device__ __forceinline__ unsigned short f2bf(float f) {
    unsigned int u = __float_as_uint(f);
    u += 0x7fffu + ((u >> 16) & 1u);   // round-to-nearest-even
    return (unsigned short)(u >> 16);
}

__device__ __forceinline__ v8f wmma_bf16(const BFrag& a, const BFrag& b, v8f c) {
    return __builtin_amdgcn_wmma_f32_16x16x32_bf16(
        false, a.v, false, b.v, (short)0, c, false, false);
}

// B fragment (32x16): mem reduction-major mat[row][k]; lane holds column `row`,
// elems 0..7 at k0+kg*8, elems 8..15 at +16 (two b128 loads)
__device__ __forceinline__ BFrag load_b(const unsigned short* mat, int row, int ld,
                                        int k0, int kg) {
    BFrag f;
    const unsigned short* p = mat + (size_t)row * ld + k0 + (kg << 3);
    f.h[0] = *(const v8us*)p;
    f.h[1] = *(const v8us*)(p + 16);
    return f;
}

// A fragment (16x32): elems 0..7 at k0+kg*8, elems 8..15 at +16
__device__ __forceinline__ BFrag load_a_bf(const unsigned short* mat, int row, int ld,
                                           int k0, int kg) {
    BFrag f;
    const unsigned short* p = mat + (size_t)row * ld + k0 + (kg << 3);
    f.h[0] = *(const v8us*)p;
    f.h[1] = *(const v8us*)(p + 16);
    return f;
}

__device__ __forceinline__ BFrag load_a_f32(const float* mat, int row, int ld,
                                            int k0, int kg) {
    const float* p = mat + (size_t)row * ld + k0 + (kg << 3);
    v4f x0 = *(const v4f*)p;
    v4f x1 = *(const v4f*)(p + 4);
    v4f y0 = *(const v4f*)(p + 16);
    v4f y1 = *(const v4f*)(p + 20);
    BFrag f;
#pragma unroll
    for (int i = 0; i < 4; ++i) {
        f.s[i]      = f2bf(x0[i]);
        f.s[4 + i]  = f2bf(x1[i]);
        f.s[8 + i]  = f2bf(y0[i]);
        f.s[12 + i] = f2bf(y1[i]);
    }
    return f;
}

// ---------- kernel 1: weight f32 -> bf16 ----------
__global__ void cvt_f32_bf16(const float* __restrict__ src,
                             unsigned short* __restrict__ dst, int n) {
    int i = blockIdx.x * blockDim.x + threadIdx.x;
    if (i < n) dst[i] = f2bf(src[i]);
}

// ---------- kernel 2: GEMM out = X @ W^T + bias, 32x64 tile per wave ----------
// MODE 0: out bf16 [B,H,S,64] | MODE 1: out bf16 [B,H,64,S] | MODE 2: out f32 [M,1024]
template <int MODE, typename InT>
__global__ void __launch_bounds__(256)
proj_gemm(const InT* __restrict__ X, const unsigned short* __restrict__ W,
          const float* __restrict__ bias, void* __restrict__ outp) {
    const int lane = threadIdx.x & 31;
    const int wid  = blockIdx.x * 8 + (threadIdx.x >> 5);  // 0..2047
    const int m0 = (wid >> 4) << 5;    // 128 tiles of 32 rows
    const int n0 = (wid & 15) << 6;    // 16 tiles of 64 cols
    const int ln = lane & 15;
    const int kg = lane >> 4;

    v8f acc[2][4] = {};
    for (int k0 = 0; k0 < 1024; k0 += 32) {
        BFrag a0, a1;
        if constexpr (sizeof(InT) == 4) {
            a0 = load_a_f32((const float*)X, m0 + ln,      1024, k0, kg);
            a1 = load_a_f32((const float*)X, m0 + 16 + ln, 1024, k0, kg);
        } else {
            a0 = load_a_bf((const unsigned short*)X, m0 + ln,      1024, k0, kg);
            a1 = load_a_bf((const unsigned short*)X, m0 + 16 + ln, 1024, k0, kg);
        }
        // prefetch all 4 B fragments, then issue 8 back-to-back WMMAs
        BFrag bfr[4];
#pragma unroll
        for (int t = 0; t < 4; ++t)
            bfr[t] = load_b(W, n0 + t * 16 + ln, 1024, k0, kg);
#pragma unroll
        for (int t = 0; t < 4; ++t) {
            acc[0][t] = wmma_bf16(a0, bfr[t], acc[0][t]);
            acc[1][t] = wmma_bf16(a1, bfr[t], acc[1][t]);
        }
    }

#pragma unroll
    for (int r = 0; r < 2; ++r) {
#pragma unroll
        for (int t = 0; t < 4; ++t) {
            const int n = n0 + t * 16 + ln;
            const float bvv = bias[n];
#pragma unroll
            for (int j = 0; j < 8; ++j) {
                const int m = m0 + r * 16 + j + (kg << 3);
                const float val = acc[r][t][j] + bvv;
                if constexpr (MODE == 0) {
                    const int bb = m >> 11, s = m & 2047, hh = n >> 6, dd = n & 63;
                    ((unsigned short*)outp)[((((size_t)bb * 16 + hh) * 2048 + s) << 6) + dd] = f2bf(val);
                } else if constexpr (MODE == 1) {
                    const int bb = m >> 11, s = m & 2047, hh = n >> 6, dd = n & 63;
                    ((unsigned short*)outp)[((((size_t)bb * 16 + hh) * 64 + dd) << 11) + s] = f2bf(val);
                } else {
                    ((float*)outp)[(size_t)m * 1024 + n] = val;
                }
            }
        }
    }
}

// ---------- kernel 3: flash attention ----------
// 4 waves / block; each wave owns 16 q-rows; block shares 64x64 K/V LDS tiles,
// double-buffered via async global->LDS. Row-sums done by WMMA against an
// all-ones B matrix; only the row-max needs lane reductions. B fragments are
// register-prefetched in groups of 4 so WMMAs aren't serialized on DScnt==0.
__global__ void __launch_bounds__(128)
flash_attn(const unsigned short* __restrict__ Q,   // [B,H,S,64] bf16
           const unsigned short* __restrict__ K,   // [B,H,S,64] bf16
           const unsigned short* __restrict__ VT,  // [B,H,64,S] bf16
           const int* __restrict__ mask,           // [B,S]
           unsigned short* __restrict__ ctx) {     // [B,S,1024] bf16
    __shared__ unsigned short Kt[2][64 * 64];      // 16 KB
    __shared__ unsigned short Vt[2][64 * 64];      // 16 KB
    __shared__ unsigned short Pl[4][16 * 64];      // 8 KB (per-wave P tiles)

    const int tid  = threadIdx.x;
    const int wv   = tid >> 5;
    const int lane = tid & 31;
    const int ln = lane & 15;
    const int kg = lane >> 4;
    const int wid  = blockIdx.x;       // 0..1023
    const int qt64 = wid & 31;
    const int bh   = wid >> 5;
    const int h = bh & 15;
    const int b = bh >> 4;

    const unsigned short* Qb = Q + ((size_t)bh * 2048 + qt64 * 64 + wv * 16) * 64;
    const unsigned short* Kb = K + (size_t)bh * 2048 * 64;
    const unsigned short* Vb = VT + (size_t)bh * 64 * 2048;
    const int* mb = mask + b * 2048;

    const BFrag qa0 = load_a_bf(Qb, ln, 64, 0, kg);
    const BFrag qa1 = load_a_bf(Qb, ln, 64, 32, kg);

    BFrag ones;
#pragma unroll
    for (int i = 0; i < 16; ++i) ones.s[i] = 0x3f80;   // bf16 1.0

    float mrow[8];
    v8f o[4] = {};
    v8f lacc = {};
#pragma unroll
    for (int j = 0; j < 8; ++j) mrow[j] = -3.0e38f;

    const float inv_scale = 0.125f;    // 1/sqrt(64)

    // stage tile 0
    {
        const unsigned short* kgp = Kb;          // 64x64 contiguous
#pragma unroll
        for (int c = 0; c < 4; ++c) {
            int idx = c * 128 + tid;             // 512 x 16B chunks
            stage16(kgp + idx * 8, &Kt[0][idx * 8]);
        }
#pragma unroll
        for (int c = 0; c < 4; ++c) {
            int idx = c * 128 + tid;
            int row = idx >> 3, col = (idx & 7) * 8;
            stage16(Vb + (size_t)row * 2048 + col, &Vt[0][row * 64 + col]);
        }
    }
    wait_async0();
    __syncthreads();

    int buf = 0;
    for (int kb = 0; kb < 2048; kb += 64) {
        // prefetch next tile into the other buffer
        if (kb + 64 < 2048) {
            const unsigned short* kgp = Kb + (size_t)(kb + 64) * 64;
#pragma unroll
            for (int c = 0; c < 4; ++c) {
                int idx = c * 128 + tid;
                stage16(kgp + idx * 8, &Kt[buf ^ 1][idx * 8]);
            }
#pragma unroll
            for (int c = 0; c < 4; ++c) {
                int idx = c * 128 + tid;
                int row = idx >> 3, col = (idx & 7) * 8;
                stage16(Vb + (size_t)row * 2048 + (kb + 64) + col,
                        &Vt[buf ^ 1][row * 64 + col]);
            }
        }

        const unsigned short* Kc = Kt[buf];
        const unsigned short* Vc = Vt[buf];

        int mk[4];
#pragma unroll
        for (int t = 0; t < 4; ++t) mk[t] = mb[kb + t * 16 + ln];

        // scores: prefetch K fragments per 32-K group, then batch the WMMAs
        v8f s[4];
        const v8f zero = {};
        BFrag kf0[4], kf1[4];
#pragma unroll
        for (int t = 0; t < 4; ++t) kf0[t] = load_b(Kc, t * 16 + ln, 64, 0, kg);
#pragma unroll
        for (int t = 0; t < 4; ++t) kf1[t] = load_b(Kc, t * 16 + ln, 64, 32, kg);
#pragma unroll
        for (int t = 0; t < 4; ++t) s[t] = wmma_bf16(qa0, kf0[t], zero);
#pragma unroll
        for (int t = 0; t < 4; ++t) s[t] = wmma_bf16(qa1, kf1[t], s[t]);

        float alpha[8];
#pragma unroll
        for (int j = 0; j < 8; ++j) {
            float a0 = s[0][j] * inv_scale; if (mk[0] == 0) a0 = -1.0e9f;
            float a1 = s[1][j] * inv_scale; if (mk[1] == 0) a1 = -1.0e9f;
            float a2 = s[2][j] * inv_scale; if (mk[2] == 0) a2 = -1.0e9f;
            float a3 = s[3][j] * inv_scale; if (mk[3] == 0) a3 = -1.0e9f;
            float tm = fmaxf(fmaxf(a0, a1), fmaxf(a2, a3));
#pragma unroll
            for (int d = 1; d < 16; d <<= 1) tm = fmaxf(tm, __shfl_xor(tm, d, 16));
            const float nm = fmaxf(mrow[j], tm);
            alpha[j] = __expf(mrow[j] - nm);
            mrow[j] = nm;
            s[0][j] = __expf(a0 - nm);
            s[1][j] = __expf(a1 - nm);
            s[2][j] = __expf(a2 - nm);
            s[3][j] = __expf(a3 - nm);
        }
#pragma unroll
        for (int t = 0; t < 4; ++t)
#pragma unroll
            for (int j = 0; j < 8; ++j) o[t][j] *= alpha[j];
#pragma unroll
        for (int j = 0; j < 8; ++j) lacc[j] *= alpha[j];

        // C-layout P -> per-wave LDS 16x64 bf16 (A-layout re-shape)
        unsigned short* Pw = Pl[wv];
#pragma unroll
        for (int t = 0; t < 4; ++t)
#pragma unroll
            for (int j = 0; j < 8; ++j)
                Pw[(j + (kg << 3)) * 64 + t * 16 + ln] = f2bf(s[t][j]);

        BFrag pa0, pa1;
        {
            const unsigned short* pp = Pw + ln * 64 + (kg << 3);
            pa0.h[0] = *(const v8us*)pp;
            pa0.h[1] = *(const v8us*)(pp + 16);
            pa1.h[0] = *(const v8us*)(pp + 32);
            pa1.h[1] = *(const v8us*)(pp + 48);
        }

        // O += P @ V ; l += P @ ones (row-sum via matrix unit)
        BFrag vf0[4], vf1[4];
#pragma unroll
        for (int t = 0; t < 4; ++t) vf0[t] = load_b(Vc, t * 16 + ln, 64, 0, kg);
#pragma unroll
        for (int t = 0; t < 4; ++t) vf1[t] = load_b(Vc, t * 16 + ln, 64, 32, kg);
#pragma unroll
        for (int t = 0; t < 4; ++t) o[t] = wmma_bf16(pa0, vf0[t], o[t]);
#pragma unroll
        for (int t = 0; t < 4; ++t) o[t] = wmma_bf16(pa1, vf1[t], o[t]);
        lacc = wmma_bf16(pa0, ones, lacc);
        lacc = wmma_bf16(pa1, ones, lacc);

        wait_async0();       // next buffer resident
        __syncthreads();     // all waves done reading current buffer
        buf ^= 1;
    }

    float rl[8];
#pragma unroll
    for (int j = 0; j < 8; ++j) {
        float l = lacc[j];
        l = (l > 0.f) ? l : 1.f;
        rl[j] = __builtin_amdgcn_rcpf(l);
    }
#pragma unroll
    for (int t = 0; t < 4; ++t) {
#pragma unroll
        for (int j = 0; j < 8; ++j) {
            const int srow = qt64 * 64 + wv * 16 + j + (kg << 3);
            const int col  = h * 64 + t * 16 + ln;
            ctx[((size_t)(b * 2048 + srow)) * 1024 + col] = f2bf(o[t][j] * rl[j]);
        }
    }
}

// ---------- launcher ----------
extern "C" void kernel_launch(void* const* d_in, const int* in_sizes, int n_in,
                              void* d_out, int out_size, void* d_ws, size_t ws_size,
                              hipStream_t stream) {
    const float* query = (const float*)d_in[0];
    const float* key_  = (const float*)d_in[1];
    const float* value = (const float*)d_in[2];
    const int*   mask  = (const int*)d_in[3];
    const float* Wq = (const float*)d_in[4];
    const float* bq = (const float*)d_in[5];
    const float* Wk = (const float*)d_in[6];
    const float* bk = (const float*)d_in[7];
    const float* Wv = (const float*)d_in[8];
    const float* bv = (const float*)d_in[9];
    const float* Wo = (const float*)d_in[10];
    const float* bo = (const float*)d_in[11];

    unsigned short* ws = (unsigned short*)d_ws;
    const size_t WELEM = (size_t)1024 * 1024;
    const size_t QELEM = (size_t)2 * 16 * 2048 * 64;
    unsigned short* Wqb = ws;
    unsigned short* Wkb = Wqb + WELEM;
    unsigned short* Wvb = Wkb + WELEM;
    unsigned short* Wob = Wvb + WELEM;
    unsigned short* Qb  = Wob + WELEM;
    unsigned short* Kb  = Qb + QELEM;
    unsigned short* VTb = Kb + QELEM;
    unsigned short* ctx = VTb + QELEM;   // ~40 MB total workspace

    cvt_f32_bf16<<<4096, 256, 0, stream>>>(Wq, Wqb, 1 << 20);
    cvt_f32_bf16<<<4096, 256, 0, stream>>>(Wk, Wkb, 1 << 20);
    cvt_f32_bf16<<<4096, 256, 0, stream>>>(Wv, Wvb, 1 << 20);
    cvt_f32_bf16<<<4096, 256, 0, stream>>>(Wo, Wob, 1 << 20);

    proj_gemm<0, float><<<256, 256, 0, stream>>>(query, Wqb, bq, Qb);
    proj_gemm<0, float><<<256, 256, 0, stream>>>(key_,  Wkb, bk, Kb);
    proj_gemm<1, float><<<256, 256, 0, stream>>>(value, Wvb, bv, VTb);

    flash_attn<<<1024, 128, 0, stream>>>(Qb, Kb, VTb, mask, ctx);

    proj_gemm<2, unsigned short><<<256, 256, 0, stream>>>(ctx, Wob, bo, d_out);
}
